// VectorQuantizer_1005022347822
// MI455X (gfx1250) — matrix operand order repaired and checked by reading, not compiled, not run on hardware
//
#include <hip/hip_runtime.h>
#include <hip/hip_bf16.h>

// ---------------------------------------------------------------------------
// VQ: distances via bf16 WMMA (f32 accumulate), argmin, gather, loss.
//   d_in[0] = inputs    [16,4096,64] f32
//   d_in[1] = embedding [1024,64]    f32
//   d_out   = quantized (4194304 f32) | loss (1 f32) | indices (65536 f32)
//   d_ws    = bf16 embedding copy | e-norms | per-block loss partials
// Embedding (bf16) + norms staged in LDS per workgroup via
// GLOBAL_LOAD_ASYNC_TO_LDS_B128 (ASYNCcnt); each wave computes a 32x1024
// distance tile = 4 WMMAs per B-fragment load.
// ---------------------------------------------------------------------------

#define NUM_EMB   1024
#define EMB_DIM   64
#define ROWS      (16 * 4096)            // 65536 flattened rows
#define Q_ELEMS   (ROWS * EMB_DIM)       // 4194304
#define WAVES_PER_BLOCK 8
#define ROWS_PER_WAVE   32               // two 16-row WMMA tiles per wave
#define ROWS_PER_BLOCK  (WAVES_PER_BLOCK * ROWS_PER_WAVE)   // 256
#define NBLOCKS   (ROWS / ROWS_PER_BLOCK)                   // 256

// workspace layout (bytes) -- ebf and enorm contiguous so one LDS copy works
#define EBF_BYTES   (NUM_EMB * EMB_DIM * 2)   // 131072
#define ENORM_BYTES (NUM_EMB * 4)             // 4096
#define LDS_BYTES   (EBF_BYTES + ENORM_BYTES) // 135168
#define PART_OFF    LDS_BYTES                 // partials after staged region

typedef __attribute__((ext_vector_type(16))) __bf16 v16bf;
typedef __attribute__((ext_vector_type(8)))  __bf16 v8bf;
typedef __attribute__((ext_vector_type(8)))  float  v8f;
typedef __attribute__((ext_vector_type(4)))  int    v4i;

typedef __attribute__((address_space(1))) v4i* gv4i_ptr;   // global <4 x i32>*
typedef __attribute__((address_space(3))) v4i* lv4i_ptr;   // LDS    <4 x i32>*

#if defined(__has_builtin)
#if __has_builtin(__builtin_amdgcn_global_load_async_to_lds_b128) && \
    __has_builtin(__builtin_amdgcn_s_wait_asynccnt)
#define VQ_ASYNC_COPY 1
#endif
#endif

// --------------------------- prep: bf16 copy + norms ------------------------
__global__ __launch_bounds__(64) void vq_prep(const float* __restrict__ emb,
                                              __bf16* __restrict__ ebf,
                                              float* __restrict__ enorm) {
  const int k = blockIdx.x;          // code id
  const int t = threadIdx.x;         // dim id (64 threads = 2 waves)
  float v = emb[(size_t)k * EMB_DIM + t];
  ebf[(size_t)k * EMB_DIM + t] = (__bf16)v;
  float s = v * v;
#pragma unroll
  for (int mk = 16; mk >= 1; mk >>= 1) s += __shfl_xor(s, mk, 32);
  __shared__ float part[2];
  if ((t & 31) == 0) part[t >> 5] = s;
  __syncthreads();
  if (t == 0) enorm[k] = part[0] + part[1];
}

// A-fragment: 16x32 bf16, lane = (M = lane&15, K-half = lane>>4)
// VGPR0-3: K = h*8 + 0..7 ; VGPR4-7: K = 16 + h*8 + 0..7   (ISA 7.12.2)
__device__ inline v16bf load_a_frag(const float* __restrict__ row, int h, int kbase) {
  const float* p0 = row + kbase + h * 8;
  const float* p1 = row + kbase + 16 + h * 8;
  float4 c0 = *(const float4*)(p0);
  float4 c1 = *(const float4*)(p0 + 4);
  float4 c2 = *(const float4*)(p1);
  float4 c3 = *(const float4*)(p1 + 4);
  v16bf a;
  a[0]  = (__bf16)c0.x; a[1]  = (__bf16)c0.y; a[2]  = (__bf16)c0.z; a[3]  = (__bf16)c0.w;
  a[4]  = (__bf16)c1.x; a[5]  = (__bf16)c1.y; a[6]  = (__bf16)c1.z; a[7]  = (__bf16)c1.w;
  a[8]  = (__bf16)c2.x; a[9]  = (__bf16)c2.y; a[10] = (__bf16)c2.z; a[11] = (__bf16)c2.w;
  a[12] = (__bf16)c3.x; a[13] = (__bf16)c3.y; a[14] = (__bf16)c3.z; a[15] = (__bf16)c3.w;
  return a;
}

// B-fragment: 32x16 bf16, lane = (N = lane&15, K-group = lane>>4)
// K = h*8 + 0..7 and 16 + h*8 + 0..7 per lane => 2 x ds_load_b128 from the LDS code row.
__device__ inline v16bf load_b_frag(const __bf16* brow, int h, int kbase) {
  const v8bf* p0 = (const v8bf*)(brow + kbase + h * 8);
  const v8bf* p1 = (const v8bf*)(brow + kbase + 16 + h * 8);
  v8bf b0 = *p0;
  v8bf b1 = *p1;
  v16bf b;
#pragma unroll
  for (int j = 0; j < 8; ++j) { b[j] = b0[j]; b[8 + j] = b1[j]; }
  return b;
}

// ------------------------------ main kernel --------------------------------
__global__ __launch_bounds__(256) void vq_main(const float* __restrict__ inp,
                                               const float* __restrict__ emb,
                                               const char* __restrict__ gstage,
                                               float* __restrict__ outq,
                                               float* __restrict__ outidx,
                                               float* __restrict__ partials) {
  extern __shared__ char smem[];
  __bf16* sB = (__bf16*)smem;                 // bf16 embedding [1024][64]
  float*  sN = (float*)(smem + EBF_BYTES);    // norms [1024]

  // ---- stage embedding (bf16) + norms into LDS: 135168 B, 33 b128/thread ----
  {
    const int tid = threadIdx.x;
#if VQ_ASYNC_COPY
    for (int off = tid * 16; off < LDS_BYTES; off += 256 * 16) {
      __builtin_amdgcn_global_load_async_to_lds_b128(
          (gv4i_ptr)(gstage + off), (lv4i_ptr)(smem + off), 0, 0);
    }
    __builtin_amdgcn_s_wait_asynccnt(0);
#else
    for (int off = tid * 16; off < LDS_BYTES; off += 256 * 16) {
      *(uint4*)(smem + off) = *(const uint4*)(gstage + off);
    }
#endif
    __syncthreads();
  }

  const int lane = threadIdx.x & 31;
  const int wave = threadIdx.x >> 5;
  const int m0   = blockIdx.x * ROWS_PER_BLOCK + wave * ROWS_PER_WAVE;
  const int m    = lane & 15;
  const int h    = lane >> 4;

  // ---- phase 1: distances via WMMA, running argmin --------------------------
  v16bf a[2][2];
#pragma unroll
  for (int t = 0; t < 2; ++t) {
    const float* arow = inp + (size_t)(m0 + t * 16 + m) * EMB_DIM;
    a[t][0] = load_a_frag(arow, h, 0);    // K = 0..31
    a[t][1] = load_a_frag(arow, h, 32);   // K = 32..63
  }

  float bd[2][8];
  int   bi[2][8];
#pragma unroll
  for (int t = 0; t < 2; ++t)
#pragma unroll
    for (int i = 0; i < 8; ++i) { bd[t][i] = 3.4e38f; bi[t][i] = 0x7fffffff; }

  for (int n0 = 0; n0 < NUM_EMB; n0 += 16) {
    const __bf16* brow = sB + (n0 + m) * EMB_DIM;
    const v16bf b0 = load_b_frag(brow, h, 0);
    const v16bf b1 = load_b_frag(brow, h, 32);
    const float en = sN[n0 + m];           // this lane's column norm
    const int nidx = n0 + m;

#pragma unroll
    for (int t = 0; t < 2; ++t) {          // 4 WMMAs per B-fragment pair
      v8f c = {};
      c = __builtin_amdgcn_wmma_f32_16x16x32_bf16(false, a[t][0], false, b0, (short)0, c, false, false);
      c = __builtin_amdgcn_wmma_f32_16x16x32_bf16(false, a[t][1], false, b1, (short)0, c, false, false);
#pragma unroll
      for (int i = 0; i < 8; ++i) {        // c[i]: row M = i + 8*h, col N = m
        float d = fmaf(-2.0f, c[i], en);   // ||e||^2 - 2 x.e  (||x||^2 const)
        if (d < bd[t][i]) { bd[t][i] = d; bi[t][i] = nidx; }  // strict <: first tie
      }
    }
  }

  // ---- argmin reduction, index writes, gather + loss per 16-row tile --------
  float acc = 0.0f;
#pragma unroll
  for (int t = 0; t < 2; ++t) {
    // cross-lane argmin within each 16-lane M-group (xor masks stay in-group)
#pragma unroll
    for (int mk = 1; mk < 16; mk <<= 1) {
#pragma unroll
      for (int i = 0; i < 8; ++i) {
        float od = __shfl_xor(bd[t][i], mk, 32);
        int   oi = __shfl_xor(bi[t][i], mk, 32);
        if (od < bd[t][i] || (od == bd[t][i] && oi < bi[t][i])) {
          bd[t][i] = od; bi[t][i] = oi;
        }
      }
    }

    // indices (as float, per output dtype): rows M = i + 8*h from lanes 0/16
    if (m == 0) {
#pragma unroll
      for (int i = 0; i < 8; ++i)
        outidx[m0 + t * 16 + h * 8 + i] = (float)bi[t][i];
    }

    // broadcast each row's index to all lanes (constant register indices)
    int idxRow[16];
#pragma unroll
    for (int r = 0; r < 16; ++r)
      idxRow[r] = __shfl(bi[t][r & 7], (r >> 3) * 16, 32);

    // gather quantized rows from f32 embedding, accumulate loss
#pragma unroll
    for (int r = 0; r < 16; ++r) {
      const int row = m0 + t * 16 + r;
      const float2 q = ((const float2*)(emb + (size_t)idxRow[r] * EMB_DIM))[lane];
      const float2 x = ((const float2*)(inp + (size_t)row * EMB_DIM))[lane];
      ((float2*)(outq + (size_t)row * EMB_DIM))[lane] = q;
      float dx = q.x - x.x, dy = q.y - x.y;
      acc = fmaf(dx, dx, acc);
      acc = fmaf(dy, dy, acc);
    }
  }

  // deterministic block reduction: wave shuffle -> LDS -> thread 0
#pragma unroll
  for (int mk = 16; mk >= 1; mk >>= 1) acc += __shfl_xor(acc, mk, 32);
  __shared__ float wsum[WAVES_PER_BLOCK];
  if (lane == 0) wsum[wave] = acc;
  __syncthreads();
  if (threadIdx.x == 0) {
    float s = 0.0f;
#pragma unroll
    for (int w = 0; w < WAVES_PER_BLOCK; ++w) s += wsum[w];
    partials[blockIdx.x] = s;
  }
}

// ------------------------------ finalize -----------------------------------
__global__ void vq_finalize(const float* __restrict__ partials,
                            float* __restrict__ out_loss) {
  if (threadIdx.x == 0 && blockIdx.x == 0) {
    float s = 0.0f;
    for (int i = 0; i < NBLOCKS; ++i) s += partials[i];   // fixed order: deterministic
    *out_loss = 0.25f * s / (float)Q_ELEMS;
  }
}

// ------------------------------ launcher -----------------------------------
extern "C" void kernel_launch(void* const* d_in, const int* in_sizes, int n_in,
                              void* d_out, int out_size, void* d_ws, size_t ws_size,
                              hipStream_t stream) {
  const float* inp = (const float*)d_in[0];   // [16,4096,64]
  const float* emb = (const float*)d_in[1];   // [1024,64]
  float* out = (float*)d_out;

  char* ws = (char*)d_ws;
  __bf16* ebf     = (__bf16*)(ws);                 // 131072 B
  float*  enorm   = (float*)(ws + EBF_BYTES);      // 4096 B (contiguous with ebf)
  float*  partial = (float*)(ws + PART_OFF);       // NBLOCKS * 4 B

  float* out_q    = out;                      // 4194304 floats
  float* out_loss = out + Q_ELEMS;            // 1 float
  float* out_idx  = out + Q_ELEMS + 1;        // 65536 floats

  hipLaunchKernelGGL(vq_prep, dim3(NUM_EMB), dim3(64), 0, stream, emb, ebf, enorm);
  hipLaunchKernelGGL(vq_main, dim3(NBLOCKS), dim3(256), LDS_BYTES, stream,
                     inp, emb, ws, out_q, out_idx, partial);
  hipLaunchKernelGGL(vq_finalize, dim3(1), dim3(1), 0, stream, partial, out_loss);
}
